// PointNet2Patchlets_44435731644682
// MI455X (gfx1250) — compile-verified, matrix operand order checked by reading, not run on hardware
//
#include <hip/hip_runtime.h>
#include <hip/hip_bf16.h>

// ---------------------------------------------------------------------------
// Types for CDNA5 WMMA (wave32): D(16x16,f32) = A(16x32,bf16) x B(32x16,bf16) + C
// ---------------------------------------------------------------------------
typedef __attribute__((ext_vector_type(16))) __bf16 v16bf;
typedef __attribute__((ext_vector_type(8)))  float  v8f;

static __device__ __forceinline__ unsigned short f2bf(float f) {
    unsigned u = __float_as_uint(f);
    unsigned r = u + 0x7FFFu + ((u >> 16) & 1u);   // round-to-nearest-even
    return (unsigned short)(r >> 16);
}
static __device__ __forceinline__ float bf2f(unsigned short h) {
    return __uint_as_float(((unsigned)h) << 16);
}
static __device__ __forceinline__ __bf16 bf_raw(unsigned short r) {
    __bf16 x; __builtin_memcpy(&x, &r, 2); return x;
}

// ---------------------------------------------------------------------------
// Generic implicit-im2col GEMM, fused bias+BN+ReLU epilogue.
//   A : bf16 weights, zero-padded [Mpad x Kpad] (Mpad=ceil32, Kpad=ceil32) ->
//       A fragments are aligned 32B vector loads, branch-free + prefetch.
//   X : bf16 activations [Cin x D x H x W] (P = D*H*W, p = (d*H+h)*W+w)
// Conv kernel (1,KH,KW), SAME padding over (H,W). KH*KW is a compile-time
// power of two, so the k -> (cin,dh,dw) split is shifts/ands.
// Per-tap index uses  ix = cin*P + base + dh*W + dw  with base/valid-ranges
// hoisted out of the K-loop.
// B gather is two-phase: phase 1 issues all 32 tap loads into distinct
// registers (s_clause batches, staggered waits -> full MLP), phase 2 packs
// fragments with masked selects.
// 2x2 register tiling per wave: two m-tiles x two p-tiles share BOTH operands
// -> 4 v_wmma per K-step from 2 A loads + 2 B gathers, halving the expensive
// gather cost per output vs a 1x2 scheme. 4 waves/block.
// ---------------------------------------------------------------------------
template <int KH, int KW>
__global__ __launch_bounds__(128)
void wmma_gemm_conv(const unsigned short* __restrict__ A,
                    const unsigned short* __restrict__ X,
                    unsigned short* __restrict__ Y,
                    float* __restrict__ Yf,
                    const float* __restrict__ bias,
                    const float* __restrict__ bng,
                    const float* __restrict__ bnb,
                    const float* __restrict__ bnm,
                    const float* __restrict__ bnv,
                    int M, int Cin, int D, int H, int W,
                    int Kpad, int padH, int padW, int relu)
{
    const int lane = threadIdx.x & 31;
    const int wv   = threadIdx.x >> 5;
    const int col  = lane & 15;
    const int half = lane >> 4;
    const int m0 = blockIdx.y * 32;                  // this wave: rows m0, m0+16
    const int p0 = (blockIdx.x * 8 + wv * 2) * 16;   // this wave: tiles p0, p0+16
    const int K   = Cin * KH * KW;
    const int P   = D * H * W;
    const int HW  = H * W;
    constexpr int KHW = KH * KW;                     // power of two here

    v8f acc00 = {}, acc01 = {}, acc10 = {}, acc11 = {};

    const unsigned short* arow0p = A + (size_t)(m0 + col)      * Kpad;  // < Mpad
    const unsigned short* arow1p = A + (size_t)(m0 + 16 + col) * Kpad;  // < Mpad

    // ---- per-tile constants hoisted out of the K-loop ----
    const int bp0 = p0 + col, bp1 = bp0 + 16;
    const bool okp0 = (bp0 < P), okp1 = (bp1 < P);
    int base0 = 0, dhlo0 = 0, dhhi0 = 0, dwlo0 = 0, dwhi0 = 0;
    int base1 = 0, dhlo1 = 0, dhhi1 = 0, dwlo1 = 0, dwhi1 = 0;
    if (okp0) {
        int d = bp0 / HW; int r = bp0 - d * HW; int h = r / W; int w = r - h * W;
        base0 = (d * H + h - padH) * W + (w - padW);
        dhlo0 = padH - h; dhhi0 = H - 1 - h + padH;
        dwlo0 = padW - w; dwhi0 = W - 1 - w + padW;
    }
    if (okp1) {
        int d = bp1 / HW; int r = bp1 - d * HW; int h = r / W; int w = r - h * W;
        base1 = (d * H + h - padH) * W + (w - padW);
        dhlo1 = padH - h; dhhi1 = H - 1 - h + padH;
        dwlo1 = padW - w; dwhi1 = W - 1 - w + padW;
    }

    for (int k0 = 0; k0 < K; k0 += 32) {
        const int kb = k0 + half * 16;

        // ---- A fragments: two aligned 32B vector loads + prefetch ----
        v16bf a0 = *(const v16bf*)(arow0p + kb);
        v16bf a1 = *(const v16bf*)(arow1p + kb);
        __builtin_prefetch(arow0p + kb + 32, 0, 0);  // speculative, fault-safe
        __builtin_prefetch(arow1p + kb + 32, 0, 0);

        // ---- phase 1: issue all 32 tap loads into distinct registers ----
        unsigned short raw0[16], raw1[16];
        bool ok0a[16], ok1a[16];
#pragma unroll
        for (int i = 0; i < 16; i++) {
            const int kk  = kb + i;
            const int cin = kk / KHW;                // shifts (KHW pow2)
            const int r   = kk - cin * KHW;
            const int dh  = r / KW;
            const int dw  = r - dh * KW;
            const bool kok = (kk < K);
            const int cofs = cin * P + dh * W + dw;  // one v_mad + adds
            const bool oa = okp0 & kok & (dh >= dhlo0) & (dh <= dhhi0)
                                       & (dw >= dwlo0) & (dw <= dwhi0);
            const bool ob = okp1 & kok & (dh >= dhlo1) & (dh <= dhhi1)
                                       & (dw >= dwlo1) & (dw <= dwhi1);
            ok0a[i] = oa;
            ok1a[i] = ob;
            raw0[i] = X[oa ? (cofs + base0) : 0];
            raw1[i] = X[ob ? (cofs + base1) : 0];
        }

        // ---- phase 2: pack fragments with masked selects ----
        v16bf b0 = {}, b1 = {};
#pragma unroll
        for (int i = 0; i < 16; i++) {
            b0[i] = bf_raw(ok0a[i] ? raw0[i] : (unsigned short)0);
            b1[i] = bf_raw(ok1a[i] ? raw1[i] : (unsigned short)0);
        }

        // ---- 2x2 tile: 4 WMMA sharing both operand fragments ----
        acc00 = __builtin_amdgcn_wmma_f32_16x16x32_bf16(
                    false, a0, false, b0, (short)0, acc00, false, false);
        acc01 = __builtin_amdgcn_wmma_f32_16x16x32_bf16(
                    false, a0, false, b1, (short)0, acc01, false, false);
        acc10 = __builtin_amdgcn_wmma_f32_16x16x32_bf16(
                    false, a1, false, b0, (short)0, acc10, false, false);
        acc11 = __builtin_amdgcn_wmma_f32_16x16x32_bf16(
                    false, a1, false, b1, (short)0, acc11, false, false);
    }

    // ---- fused epilogue: bias + BN + ReLU, dual-precision store ----
#pragma unroll
    for (int mt = 0; mt < 2; mt++) {
#pragma unroll
        for (int r = 0; r < 8; r++) {
            const int m = m0 + mt * 16 + r + half * 8;
            if (m >= M) continue;
            float bia = bias ? bias[m] : 0.f;
            float sc = 1.f, sh = 0.f;
            if (bng) { sc = bng[m] * rsqrtf(bnv[m] + 1e-5f); sh = bnb[m] - bnm[m] * sc; }
            float v0 = (mt == 0) ? acc00[r] : acc10[r];
            float v1 = (mt == 0) ? acc01[r] : acc11[r];
            if (bp0 < P) {
                float v = (v0 + bia) * sc + sh;
                if (relu) v = fmaxf(v, 0.f);
                if (Y)  Y[(size_t)m * P + bp0]  = f2bf(v);
                if (Yf) Yf[(size_t)m * P + bp0] = v;
            }
            if (bp1 < P) {
                float v = (v1 + bia) * sc + sh;
                if (relu) v = fmaxf(v, 0.f);
                if (Y)  Y[(size_t)m * P + bp1]  = f2bf(v);
                if (Yf) Yf[(size_t)m * P + bp1] = v;
            }
        }
    }
}

// ---------------------------------------------------------------------------
// f32 [M x K] -> bf16 zero-padded [Mpad x Kpad]
// ---------------------------------------------------------------------------
__global__ void f32_to_bf16_pad_kernel(const float* __restrict__ s,
                                       unsigned short* __restrict__ d,
                                       int M, int K, int Mpad, int Kpad)
{
    int i = blockIdx.x * blockDim.x + threadIdx.x;
    if (i >= Mpad * Kpad) return;
    int m = i / Kpad, k = i - m * Kpad;
    d[i] = (m < M && k < K) ? f2bf(s[(size_t)m * K + k]) : (unsigned short)0;
}

// ---------------------------------------------------------------------------
// Per-frame 16-NN of prev-frame query points in current frame (top_k of -d2).
// xyz layout: [T][3][N] row-major.  idxs: [T][N][16]
// ---------------------------------------------------------------------------
__global__ void knn16_kernel(const float* __restrict__ xyz,
                             int* __restrict__ idxs, int T, int N)
{
    int gid = blockIdx.x * blockDim.x + threadIdx.x;
    if (gid >= T * N) return;
    int f = gid / N, q = gid - f * N;
    int fp = (f == 0) ? 0 : (f - 1);
    const float* cur = xyz + (size_t)f  * 3 * N;
    const float* prv = xyz + (size_t)fp * 3 * N;
    float qx = prv[q], qy = prv[N + q], qz = prv[2 * N + q];

    float bd[16]; int bi[16];
#pragma unroll
    for (int i = 0; i < 16; i++) { bd[i] = 3.4e38f; bi[i] = 0; }

    for (int j = 0; j < N; j++) {
        float dx = cur[j] - qx, dy = cur[N + j] - qy, dz = cur[2 * N + j] - qz;
        float dd = dx * dx + dy * dy + dz * dz;
        if (dd < bd[15]) {
            int pos = 15;
            while (pos > 0 && bd[pos - 1] > dd) pos--;
            for (int s = 15; s > pos; s--) { bd[s] = bd[s - 1]; bi[s] = bi[s - 1]; }
            bd[pos] = dd; bi[pos] = j;
        }
    }
    int* o = idxs + (size_t)gid * 16;
#pragma unroll
    for (int i = 0; i < 16; i++) o[i] = bi[i];
}

// patchlets[f] = (f==0) ? idxs[0] : idxs[f][ patchlets[f-1][:,0] ]  (scan)
__global__ void chain_kernel(const int* __restrict__ idxs,
                             int* __restrict__ patch, int f, int N)
{
    int q = blockIdx.x * blockDim.x + threadIdx.x;
    if (q >= N) return;
    if (f == 0) {
        for (int k = 0; k < 16; k++)
            patch[(size_t)q * 16 + k] = idxs[(size_t)q * 16 + k];
    } else {
        int prev0 = patch[((size_t)(f - 1) * N + q) * 16 + 0];
        for (int k = 0; k < 16; k++)
            patch[((size_t)f * N + q) * 16 + k] =
                idxs[((size_t)f * N + prev0) * 16 + k];
    }
}

// ---------------------------------------------------------------------------
// Farthest-point sampling: one block per batch, dist + argmax reduction in LDS.
// ---------------------------------------------------------------------------
__global__ __launch_bounds__(256)
void fps_kernel(const float* __restrict__ xyz, long long batchStride,
                int ptStride, int cStride, int N, int npoint, int* __restrict__ out)
{
    __shared__ float dist[1024];
    __shared__ float rmax[256];
    __shared__ int   rarg[256];
    __shared__ int   far_s;
    const float* base = xyz + (size_t)blockIdx.x * batchStride;
    int tid = threadIdx.x;
    for (int i = tid; i < N; i += 256) dist[i] = 1e10f;
    if (tid == 0) far_s = 0;
    __syncthreads();

    for (int it = 0; it < npoint; it++) {
        int far = far_s;
        if (tid == 0) out[(size_t)blockIdx.x * npoint + it] = far;
        float cx = base[(size_t)far * ptStride];
        float cy = base[(size_t)far * ptStride + cStride];
        float cz = base[(size_t)far * ptStride + 2 * cStride];
        float bm = -1.f; int ba = 0;
        for (int i = tid; i < N; i += 256) {
            float dx = base[(size_t)i * ptStride]               - cx;
            float dy = base[(size_t)i * ptStride + cStride]     - cy;
            float dz = base[(size_t)i * ptStride + 2 * cStride] - cz;
            float dd = dx * dx + dy * dy + dz * dz;
            float nd = fminf(dist[i], dd);
            dist[i] = nd;
            if (nd > bm) { bm = nd; ba = i; }
        }
        rmax[tid] = bm; rarg[tid] = ba;
        __syncthreads();
        for (int s = 128; s > 0; s >>= 1) {
            if (tid < s) {
                if (rmax[tid + s] > rmax[tid] ||
                    (rmax[tid + s] == rmax[tid] && rarg[tid + s] < rarg[tid])) {
                    rmax[tid] = rmax[tid + s]; rarg[tid] = rarg[tid + s];
                }
            }
            __syncthreads();
        }
        if (tid == 0) far_s = rarg[0];
        __syncthreads();
    }
}

// ---------------------------------------------------------------------------
// Build normalized patchlet tensor Xptc[3][S*T*16] (p=(s*T+f)*16+k) and the
// unnormalized first-point coords pp0[T][S][3] used as sa1's xyz.
// ---------------------------------------------------------------------------
__global__ void gather_patchlets_kernel(const float* __restrict__ xyz,
                                        const int* __restrict__ patch,
                                        const int* __restrict__ fidx,
                                        unsigned short* __restrict__ Xp,
                                        float* __restrict__ pp0,
                                        int T, int N, int S)
{
    int gid = blockIdx.x * blockDim.x + threadIdx.x;
    int P = S * T * 16;
    if (gid >= P) return;
    int k = gid & 15;
    int f = (gid >> 4) % T;
    int s = gid / (16 * T);

    int sp = fidx[s];
    int pt = patch[((size_t)f * N + sp) * 16 + k];
    int rp = patch[(size_t)sp * 16 + 0];          // frame 0, first patch point
    const float* cf = xyz + (size_t)f * 3 * N;
    float rx = xyz[rp], ry = xyz[N + rp], rz = xyz[2 * N + rp];
    float x = cf[pt], y = cf[N + pt], z = cf[2 * N + pt];

    Xp[(size_t)0 * P + gid] = f2bf(x - rx);
    Xp[(size_t)1 * P + gid] = f2bf(y - ry);
    Xp[(size_t)2 * P + gid] = f2bf(z - rz);
    if (k == 0) {
        float* o = pp0 + ((size_t)f * S + s) * 3;
        o[0] = x; o[1] = y; o[2] = z;
    }
}

// gather fps-selected coordinates: src[T][n][3], idx[T][S] -> dst[T][S][3]
__global__ void gather_centers_kernel(const float* __restrict__ src,
                                      const int* __restrict__ idx,
                                      float* __restrict__ dst, int T, int n, int S)
{
    int gid = blockIdx.x * blockDim.x + threadIdx.x;
    if (gid >= T * S) return;
    int f = gid / S, j = gid - f * S;
    int p = idx[(size_t)f * S + j];
    const float* s0 = src + ((size_t)f * n + p) * 3;
    float* d0 = dst + ((size_t)f * S + j) * 3;
    d0[0] = s0[0]; d0[1] = s0[1]; d0[2] = s0[2];
}

// ---------------------------------------------------------------------------
// Ball-query grouping (first ns hits in index order, padded with first hit).
// X[(3+Cf)][S*T*ns]: ch 0..2 centered xyz, ch 3.. feats (feats[Cf][n*T], pt*T+f)
// ---------------------------------------------------------------------------
__global__ void sa_group_kernel(const float* __restrict__ xyzsrc,
                                const float* __restrict__ centers,
                                const unsigned short* __restrict__ feats,
                                unsigned short* __restrict__ X,
                                int T, int n, int S, int ns, int Cf, float r2)
{
    int gid = blockIdx.x * blockDim.x + threadIdx.x;
    if (gid >= T * S) return;
    int f = gid % T, j = gid / T;
    const float* xb = xyzsrc + (size_t)f * n * 3;
    const float* c  = centers + ((size_t)f * S + j) * 3;
    float cx = c[0], cy = c[1], cz = c[2];
    size_t P = (size_t)S * T * ns;
    size_t pbase = ((size_t)j * T + f) * ns;

    int cnt = 0, first = -1;
    for (int i = 0; i < n && cnt < ns; i++) {
        float dx = xb[i * 3] - cx, dy = xb[i * 3 + 1] - cy, dz = xb[i * 3 + 2] - cz;
        float dd = dx * dx + dy * dy + dz * dz;
        if (dd <= r2) {
            if (first < 0) first = i;
            size_t p = pbase + cnt;
            X[0 * P + p] = f2bf(dx);
            X[1 * P + p] = f2bf(dy);
            X[2 * P + p] = f2bf(dz);
            for (int cc = 0; cc < Cf; cc++)
                X[(size_t)(3 + cc) * P + p] =
                    feats[(size_t)cc * n * T + (size_t)i * T + f];
            cnt++;
        }
    }
    if (first < 0) first = 0;
    float fx = xb[first * 3] - cx, fy = xb[first * 3 + 1] - cy, fz = xb[first * 3 + 2] - cz;
    for (int q = cnt; q < ns; q++) {
        size_t p = pbase + q;
        X[0 * P + p] = f2bf(fx);
        X[1 * P + p] = f2bf(fy);
        X[2 * P + p] = f2bf(fz);
        for (int cc = 0; cc < Cf; cc++)
            X[(size_t)(3 + cc) * P + p] =
                feats[(size_t)cc * n * T + (size_t)first * T + f];
    }
}

// group_all (sa3): X[(3+Cf)][T*n], p = f*n+i, raw xyz + feats
__global__ void group_all_kernel(const float* __restrict__ xyzsrc,
                                 const unsigned short* __restrict__ feats,
                                 unsigned short* __restrict__ X,
                                 int T, int n, int Cf)
{
    int gid = blockIdx.x * blockDim.x + threadIdx.x;
    if (gid >= T * n) return;
    int i = gid % n, f = gid / n;
    size_t P = (size_t)T * n;
    size_t p = (size_t)f * n + i;
    const float* xb = xyzsrc + ((size_t)f * n + i) * 3;
    X[0 * P + p] = f2bf(xb[0]);
    X[1 * P + p] = f2bf(xb[1]);
    X[2 * P + p] = f2bf(xb[2]);
    for (int cc = 0; cc < Cf; cc++)
        X[(size_t)(3 + cc) * P + p] = feats[(size_t)cc * n * T + (size_t)i * T + f];
}

// pool over last dim: in[C][DH][W] -> out[C][DH] (max or mean)
__global__ void pool_kernel(const unsigned short* __restrict__ in,
                            unsigned short* __restrict__ out,
                            int C, int DH, int W, int is_max)
{
    int gid = blockIdx.x * blockDim.x + threadIdx.x;
    if (gid >= C * DH) return;
    const unsigned short* p = in + (size_t)gid * W;
    if (is_max) {
        float m = -3.4e38f;
        for (int w = 0; w < W; w++) m = fmaxf(m, bf2f(p[w]));
        out[gid] = f2bf(m);
    } else {
        float s = 0.f;
        for (int w = 0; w < W; w++) s += bf2f(p[w]);
        out[gid] = f2bf(s / (float)W);
    }
}

// l3_pts (b,t,1024): out[f*C+c] = L3[c*T+f]
__global__ void l3_out_kernel(const unsigned short* __restrict__ L3,
                              float* __restrict__ out, int C, int T)
{
    int gid = blockIdx.x * blockDim.x + threadIdx.x;
    if (gid >= C * T) return;
    int c = gid / T, f = gid - c * T;
    out[(size_t)f * C + c] = bf2f(L3[(size_t)c * T + f]);
}

// log_softmax over C=40 logits per frame column; pred[c][f] layout
__global__ void logsoftmax_kernel(const float* __restrict__ Yf,
                                  float* __restrict__ out, int C, int T)
{
    int f = blockIdx.x * blockDim.x + threadIdx.x;
    if (f >= T) return;
    float mx = -3.4e38f;
    for (int c = 0; c < C; c++) mx = fmaxf(mx, Yf[(size_t)c * T + f]);
    float s = 0.f;
    for (int c = 0; c < C; c++) s += __expf(Yf[(size_t)c * T + f] - mx);
    float ls = __logf(s);
    for (int c = 0; c < C; c++)
        out[(size_t)c * T + f] = Yf[(size_t)c * T + f] - mx - ls;
}

// ===========================================================================
// Host orchestration
// ===========================================================================
extern "C" void kernel_launch(void* const* d_in, const int* in_sizes, int n_in,
                              void* d_out, int out_size, void* d_ws, size_t ws_size,
                              hipStream_t stream)
{
    (void)in_sizes; (void)n_in; (void)out_size; (void)ws_size;
    const int T = 16, N = 1024, S0 = 512, KP = 16;

    // ---- parse inputs in setup_inputs() dict order ----
    int ci = 0;
    const float* in_xyz = (const float*)d_in[ci++];
    struct ConvP { const float* w; const float* b; };
    struct BnP   { const float* g; const float* be; const float* mu; const float* var; };
    auto getConv = [&]() { ConvP c; c.w = (const float*)d_in[ci++]; c.b = (const float*)d_in[ci++]; return c; };
    auto getBn   = [&]() { BnP b; b.g = (const float*)d_in[ci++]; b.be = (const float*)d_in[ci++];
                           b.mu = (const float*)d_in[ci++]; b.var = (const float*)d_in[ci++]; return b; };

    ConvP ptcC[3]; for (int i = 0; i < 3; i++) ptcC[i] = getConv();
    BnP   ptcB[3]; for (int i = 0; i < 3; i++) ptcB[i] = getBn();
    ConvP tconv = getConv(); BnP bnt = getBn();
    ConvP sa1C[3]; for (int i = 0; i < 3; i++) sa1C[i] = getConv();
    BnP   sa1B[3]; for (int i = 0; i < 3; i++) sa1B[i] = getBn();
    ConvP sa2C[3]; for (int i = 0; i < 3; i++) sa2C[i] = getConv();
    BnP   sa2B[3]; for (int i = 0; i < 3; i++) sa2B[i] = getBn();
    ConvP sa3C[3]; for (int i = 0; i < 3; i++) sa3C[i] = getConv();
    BnP   sa3B[3]; for (int i = 0; i < 3; i++) sa3B[i] = getBn();
    ConvP fc1 = getConv(); BnP bn1 = getBn();
    ConvP fc2 = getConv(); BnP bn2 = getBn();
    ConvP tc2 = getConv(); BnP bn3 = getBn();
    ConvP fc3 = getConv();

    // ---- workspace bump allocator ----
    char* ws = (char*)d_ws;
    size_t off = 0;
    auto alloc = [&](size_t bytes) -> void* {
        off = (off + 255) & ~(size_t)255;
        void* p = ws + off;
        off += bytes;
        return p;
    };
    // convert [M x K] f32 weights into zero-padded [Mpad x Kpad] bf16
    // (Mpad rounded to 32 so the 2x2-tiled GEMM's second m-tile reads valid rows)
    auto toBf = [&](const float* src, int M, int K) -> unsigned short* {
        int Mpad = (M + 31) & ~31, Kpad = (K + 31) & ~31;
        unsigned short* d = (unsigned short*)alloc((size_t)Mpad * Kpad * 2);
        int n = Mpad * Kpad;
        f32_to_bf16_pad_kernel<<<(unsigned)((n + 255) / 256), 256, 0, stream>>>(
            src, d, M, K, Mpad, Kpad);
        return d;
    };
    auto gemm = [&](const unsigned short* A, const unsigned short* X,
                    unsigned short* Y, float* Yf, const float* bias, const BnP* bn,
                    int M, int Cin, int D, int H, int W,
                    int KH, int KW, int padH, int padW, int relu) {
        int K = Cin * KH * KW;
        int Kpad = (K + 31) & ~31, Mpad = (M + 31) & ~31;
        dim3 g((unsigned)((D * H * W + 127) / 128), (unsigned)(Mpad / 32));
        const float* g_ = bn ? bn->g : nullptr;  const float* be_ = bn ? bn->be : nullptr;
        const float* mu_ = bn ? bn->mu : nullptr; const float* va_ = bn ? bn->var : nullptr;
#define LAUNCH_GEMM(KH_, KW_)                                                     \
        wmma_gemm_conv<KH_, KW_><<<g, 128, 0, stream>>>(                          \
            A, X, Y, Yf, bias, g_, be_, mu_, va_,                                 \
            M, Cin, D, H, W, Kpad, padH, padW, relu)
        if      (KH == 1 && KW == 1)  LAUNCH_GEMM(1, 1);
        else if (KH == 1 && KW == 8)  LAUNCH_GEMM(1, 8);
        else if (KH == 8 && KW == 8)  LAUNCH_GEMM(8, 8);
        else if (KH == 4 && KW == 8)  LAUNCH_GEMM(4, 8);
        else if (KH == 4 && KW == 1)  LAUNCH_GEMM(4, 1);
        else if (KH == 1 && KW == 16) LAUNCH_GEMM(1, 16);
#undef LAUNCH_GEMM
    };

    // ---- convert all weights to padded bf16 once per launch ----
    unsigned short* ptcW[3] = { toBf(ptcC[0].w, 64, 3), toBf(ptcC[1].w, 64, 64), toBf(ptcC[2].w, 64, 64) };
    unsigned short* tconvW  = toBf(tconv.w, 64, 64 * 8);
    unsigned short* sa1W[3] = { toBf(sa1C[0].w, 64, 67 * 64),
                                toBf(sa1C[1].w, 64, 64 * 64),
                                toBf(sa1C[2].w, 128, 64 * 64) };
    unsigned short* sa2W[3] = { toBf(sa2C[0].w, 128, 131 * 32),
                                toBf(sa2C[1].w, 128, 128 * 32),
                                toBf(sa2C[2].w, 256, 128 * 32) };
    unsigned short* sa3W[3] = { toBf(sa3C[0].w, 256, 259 * 4),
                                toBf(sa3C[1].w, 512, 256 * 4),
                                toBf(sa3C[2].w, 1024, 512 * 4) };
    unsigned short* fc1W = toBf(fc1.w, 512, 1024);
    unsigned short* fc2W = toBf(fc2.w, 256, 512);
    unsigned short* tc2W = toBf(tc2.w, 256, 256 * 16);
    unsigned short* fc3W = toBf(fc3.w, 40, 256);

    // ---- scratch buffers ----
    int*   idxs  = (int*)alloc((size_t)T * N * 16 * 4);
    int*   patch = (int*)alloc((size_t)T * N * 16 * 4);
    int*   fidx  = (int*)alloc((size_t)S0 * 4);
    int*   idx1  = (int*)alloc((size_t)T * 512 * 4);
    int*   idx2  = (int*)alloc((size_t)T * 128 * 4);
    float* pp0   = (float*)alloc((size_t)T * 512 * 3 * 4);
    float* ctr1  = (float*)alloc((size_t)T * 512 * 3 * 4);
    float* ctr2  = (float*)alloc((size_t)T * 128 * 3 * 4);
    float* Yfc3  = (float*)alloc((size_t)40 * 16 * 4);
    unsigned short* M1 = (unsigned short*)alloc((size_t)64  * 8192 * 2);
    unsigned short* F  = (unsigned short*)alloc((size_t)64  * 8192 * 2);
    unsigned short* L1 = (unsigned short*)alloc((size_t)128 * 8192 * 2);
    unsigned short* L2 = (unsigned short*)alloc((size_t)256 * 2048 * 2);
    unsigned short* L3 = (unsigned short*)alloc((size_t)1024 * 16 * 2);
    // big ping-pong pools (peak: 67x262144 for X, 128x262144 for Y)
    unsigned short* Xpool = (unsigned short*)alloc((size_t)67  * 262144 * 2);
    unsigned short* Apool = (unsigned short*)alloc((size_t)128 * 262144 * 2);
    unsigned short* Bpool = (unsigned short*)alloc((size_t)128 * 262144 * 2);

    // ================= patchlet construction =================
    knn16_kernel<<<(T * N + 255) / 256, 256, 0, stream>>>(in_xyz, idxs, T, N);
    for (int f = 0; f < T; f++)
        chain_kernel<<<(N + 255) / 256, 256, 0, stream>>>(idxs, patch, f, N);
    fps_kernel<<<1, 256, 0, stream>>>(in_xyz, 0LL, 1, N, N, S0, fidx);
    gather_patchlets_kernel<<<(S0 * T * KP + 255) / 256, 256, 0, stream>>>(
        in_xyz, patch, fidx, Xpool, pp0, T, N, S0);

    // ================= ptc: 3 pointwise convs + mean + tconv =================
    gemm(ptcW[0], Xpool, Apool, nullptr, ptcC[0].b, &ptcB[0], 64, 3,  512, 16, 16, 1, 1, 0, 0, 1);
    gemm(ptcW[1], Apool, Bpool, nullptr, ptcC[1].b, &ptcB[1], 64, 64, 512, 16, 16, 1, 1, 0, 0, 1);
    gemm(ptcW[2], Bpool, Apool, nullptr, ptcC[2].b, &ptcB[2], 64, 64, 512, 16, 16, 1, 1, 0, 0, 1);
    pool_kernel<<<(64 * 8192 + 255) / 256, 256, 0, stream>>>(Apool, M1, 64, 8192, 16, 0);
    gemm(tconvW, M1, F, nullptr, tconv.b, &bnt, 64, 64, 1, 512, 16, 1, 8, 0, 3, 1);

    // ================= sa1 (S=512, ns=32, r=0.2, conv 1x8x8) =================
    fps_kernel<<<T, 256, 0, stream>>>(pp0, 512LL * 3, 3, 1, 512, 512, idx1);
    gather_centers_kernel<<<(T * 512 + 255) / 256, 256, 0, stream>>>(pp0, idx1, ctr1, T, 512, 512);
    sa_group_kernel<<<(T * 512 + 255) / 256, 256, 0, stream>>>(
        pp0, ctr1, F, Xpool, T, 512, 512, 32, 64, 0.2f * 0.2f);
    gemm(sa1W[0], Xpool, Apool, nullptr, sa1C[0].b, &sa1B[0], 64,  67, 512, 16, 32, 8, 8, 3, 3, 1);
    gemm(sa1W[1], Apool, Bpool, nullptr, sa1C[1].b, &sa1B[1], 64,  64, 512, 16, 32, 8, 8, 3, 3, 1);
    gemm(sa1W[2], Bpool, Apool, nullptr, sa1C[2].b, &sa1B[2], 128, 64, 512, 16, 32, 8, 8, 3, 3, 1);
    pool_kernel<<<(128 * 8192 + 255) / 256, 256, 0, stream>>>(Apool, L1, 128, 8192, 32, 1);

    // ================= sa2 (S=128, ns=64, r=0.4, conv 1x4x8) =================
    fps_kernel<<<T, 256, 0, stream>>>(ctr1, 512LL * 3, 3, 1, 512, 128, idx2);
    gather_centers_kernel<<<(T * 128 + 255) / 256, 256, 0, stream>>>(ctr1, idx2, ctr2, T, 512, 128);
    sa_group_kernel<<<(T * 128 + 255) / 256, 256, 0, stream>>>(
        ctr1, ctr2, L1, Xpool, T, 512, 128, 64, 128, 0.4f * 0.4f);
    gemm(sa2W[0], Xpool, Apool, nullptr, sa2C[0].b, &sa2B[0], 128, 131, 128, 16, 64, 4, 8, 1, 3, 1);
    gemm(sa2W[1], Apool, Bpool, nullptr, sa2C[1].b, &sa2B[1], 128, 128, 128, 16, 64, 4, 8, 1, 3, 1);
    gemm(sa2W[2], Bpool, Apool, nullptr, sa2C[2].b, &sa2B[2], 256, 128, 128, 16, 64, 4, 8, 1, 3, 1);
    pool_kernel<<<(256 * 2048 + 255) / 256, 256, 0, stream>>>(Apool, L2, 256, 2048, 64, 1);

    // ================= sa3 (group_all, n=128, conv 1x4x1) =================
    group_all_kernel<<<(T * 128 + 255) / 256, 256, 0, stream>>>(ctr2, L2, Xpool, T, 128, 256);
    gemm(sa3W[0], Xpool, Apool, nullptr, sa3C[0].b, &sa3B[0], 256,  259, 1, 16, 128, 4, 1, 1, 0, 1);
    gemm(sa3W[1], Apool, Bpool, nullptr, sa3C[1].b, &sa3B[1], 512,  256, 1, 16, 128, 4, 1, 1, 0, 1);
    gemm(sa3W[2], Bpool, Apool, nullptr, sa3C[2].b, &sa3B[2], 1024, 512, 1, 16, 128, 4, 1, 1, 0, 1);
    pool_kernel<<<(1024 * 16 + 255) / 256, 256, 0, stream>>>(Apool, L3, 1024, 16, 128, 1);
    l3_out_kernel<<<(1024 * 16 + 255) / 256, 256, 0, stream>>>(L3, (float*)d_out + 640, 1024, T);

    // ================= classification head =================
    gemm(fc1W, L3,    Apool, nullptr, fc1.b, &bn1, 512, 1024, 1, 1, 16, 1, 1, 0, 0, 1);
    gemm(fc2W, Apool, Bpool, nullptr, fc2.b, &bn2, 256, 512,  1, 1, 16, 1, 1, 0, 0, 1);
    gemm(tc2W, Bpool, Apool, nullptr, tc2.b, &bn3, 256, 256,  1, 1, 16, 1, 16, 0, 7, 1);
    gemm(fc3W, Apool, nullptr, Yfc3,  fc3.b, nullptr, 40, 256, 1, 1, 16, 1, 1, 0, 0, 0);
    logsoftmax_kernel<<<1, 64, 0, stream>>>(Yfc3, (float*)d_out, 40, T);
}
